// EngramPt_2491081032039
// MI455X (gfx1250) — matrix operand order repaired and checked by reading, not compiled, not run on hardware
//
#include <hip/hip_runtime.h>

// ---------------------------------------------------------------------------
// Shapes (fixed by the reference)
// ---------------------------------------------------------------------------
#define BB   4
#define TT   4096
#define EE   1024
#define HH   1024
#define GG   4
#define MTOT (BB * TT)          // 16384 rows
#define NTOT (GG * HH + HH)     // 5120 output channels (key || v)

// GEMM tiling
#define TM   64                 // M tile per block
#define TN   128                // N tile per block
#define KC   32                 // K chunk (one 16x16x32 WMMA step)
#define NKC  (EE / KC)          // 32 K chunks
#define LDW  20                 // uints per LDS row (= 40 halves = 80 B, 16B aligned)

typedef __attribute__((ext_vector_type(16))) _Float16 v16h;
typedef __attribute__((ext_vector_type(8)))  float    v8f;
typedef int v4i_ __attribute__((vector_size(16)));   // matches builtin param type

struct U8x { uint4 lo; uint4 hi; };   // 32 bytes -> bit_cast to v16h

#if __has_builtin(__builtin_amdgcn_global_load_async_to_lds_b128) && \
    __has_builtin(__builtin_amdgcn_s_wait_asynccnt)
#define HAVE_ASYNC 1
#else
#define HAVE_ASYNC 0
#endif

#if HAVE_ASYNC
// global -> LDS direct copy, 16B per lane, tracked by ASYNCcnt
#define CPA(dst, src)                                                     \
  __builtin_amdgcn_global_load_async_to_lds_b128(                         \
      (__attribute__((address_space(1))) v4i_*)(src),                     \
      (__attribute__((address_space(3))) v4i_*)(dst), 0, 0)
#else
#define CPA(dst, src) (*(uint4*)(dst) = *(const uint4*)(src))
#endif

__device__ inline unsigned int pk2h(float a, float b) {
  _Float16 ha = (_Float16)a, hb = (_Float16)b;   // RNE f32->f16
  unsigned short ua = __builtin_bit_cast(unsigned short, ha);
  unsigned short ub = __builtin_bit_cast(unsigned short, hb);
  return (unsigned int)ua | ((unsigned int)ub << 16);
}

// ---------------------------------------------------------------------------
// Kernel 0: one-shot f32 -> f16 convert (emb, Wk, Wv). 8 elements / thread.
// ---------------------------------------------------------------------------
__global__ __launch_bounds__(256) void f32_to_f16_kernel(
    const float* __restrict__ src, unsigned int* __restrict__ dst)
{
  const size_t i = ((size_t)blockIdx.x * 256 + threadIdx.x) * 8;
  const float4 a = *(const float4*)(src + i);
  const float4 b = *(const float4*)(src + i + 4);
  uint4 p = { pk2h(a.x, a.y), pk2h(a.z, a.w), pk2h(b.x, b.y), pk2h(b.z, b.w) };
  *(uint4*)(dst + i / 2) = p;
}

// ---------------------------------------------------------------------------
// Kernel 1: fused projection GEMM (f16 WMMA, f32 accum), async double-buffered.
//   key[m, 0..4095] = emb[m,:] . Wk[n,:] + bk[n]
//   v  [m, 0..1023] = emb[m,:] . Wv[n,:] + bv[n]
// WH = [Wk ; Wv] pre-converted to f16, shape [5120, 1024].
// ---------------------------------------------------------------------------
__global__ __launch_bounds__(256) void proj_gemm_kernel(
    const unsigned short* __restrict__ embH,   // [M,1024] f16
    const unsigned short* __restrict__ WH,     // [5120,1024] f16
    const float* __restrict__ bk, const float* __restrict__ bv,
    float* __restrict__ key_out, float* __restrict__ v_out)
{
  __shared__ unsigned int sA[2][TM * LDW];   // 2 x (64 rows x 40 f16, padded)
  __shared__ unsigned int sB[2][TN * LDW];   // 2 x (128 rows x 40 f16, padded)

  const int tid  = threadIdx.x;
  const int lane = tid & 31, wave = tid >> 5;
  const int m0   = blockIdx.y * TM;
  const int n0   = blockIdx.x * TN;

  // staging: A row=tid>>2, 8 halves (16B); B row=tid>>1, two 16B segments
  const int ar = tid >> 2, as_ = tid & 3;
  const int br = tid >> 1, bs_ = (tid & 1) * 2;
  const unsigned short* aSrc = embH + (size_t)(m0 + ar) * EE + as_ * 8;
  const unsigned short* bSrc = WH   + (size_t)(n0 + br) * EE + bs_ * 8;
  unsigned int* aDst0 = &sA[0][ar * LDW + as_ * 4];
  unsigned int* bDst0 = &sB[0][br * LDW + bs_ * 4];

  auto issue = [&](int kc, int buf) {
    const int koff = kc * KC;    // in halves
    CPA(aDst0 + buf * (TM * LDW), aSrc + koff);
    CPA(bDst0 + buf * (TN * LDW), bSrc + koff);
    CPA(bDst0 + buf * (TN * LDW) + 4, bSrc + koff + 8);
  };

  // fragment offsets (ISA 7.12.2 f16 A 16x32 / B 32x16 layouts)
  const int wm = wave >> 2, wn = wave & 3;     // 2x4 wave grid over 64x128
  const int frow = lane & 15, lh = lane >> 4;
  const int oA0 = (wm * 32 + frow) * LDW + 4 * lh;   // K = 8*lh..+8, +16..
  const int oA1 = oA0 + 16 * LDW;
  const int oB0 = (wn * 32 + frow) * LDW + 4 * lh;   // K = 8*lh..+8, +16..
  const int oB1 = oB0 + 16 * LDW;

  v8f acc00 = {}, acc01 = {}, acc10 = {}, acc11 = {};

  issue(0, 0);
  for (int kc = 0; kc < NKC; ++kc) {
    const int cur = kc & 1;
    if (kc + 1 < NKC) {
      issue(kc + 1, cur ^ 1);
#if HAVE_ASYNC
      __builtin_amdgcn_s_wait_asynccnt(3);   // chunk kc landed (in-order)
#endif
    } else {
#if HAVE_ASYNC
      __builtin_amdgcn_s_wait_asynccnt(0);
#endif
    }
    __syncthreads();   // all waves' async writes for chunk kc visible

    const unsigned int* bufA = sA[cur];
    const unsigned int* bufB = sB[cur];
    U8x ua0 = { *(const uint4*)(bufA + oA0), *(const uint4*)(bufA + oA0 + 8) };
    U8x ua1 = { *(const uint4*)(bufA + oA1), *(const uint4*)(bufA + oA1 + 8) };
    U8x ub0 = { *(const uint4*)(bufB + oB0), *(const uint4*)(bufB + oB0 + 8) };
    U8x ub1 = { *(const uint4*)(bufB + oB1), *(const uint4*)(bufB + oB1 + 8) };
    v16h fa0 = __builtin_bit_cast(v16h, ua0);
    v16h fa1 = __builtin_bit_cast(v16h, ua1);
    v16h fb0 = __builtin_bit_cast(v16h, ub0);
    v16h fb1 = __builtin_bit_cast(v16h, ub1);

    acc00 = __builtin_amdgcn_wmma_f32_16x16x32_f16(false, fa0, false, fb0, (short)0, acc00, false, false);
    acc01 = __builtin_amdgcn_wmma_f32_16x16x32_f16(false, fa0, false, fb1, (short)0, acc01, false, false);
    acc10 = __builtin_amdgcn_wmma_f32_16x16x32_f16(false, fa1, false, fb0, (short)0, acc10, false, false);
    acc11 = __builtin_amdgcn_wmma_f32_16x16x32_f16(false, fa1, false, fb1, (short)0, acc11, false, false);

    __syncthreads();   // reads done before buffer `cur` is refilled
  }

  // Store: C layout = vgpr r -> M = r + 8*(lane>>4), N = lane&15
  const bool isKey = (n0 < GG * HH);   // tiles never straddle (4096 % 128 == 0)
#pragma unroll
  for (int i = 0; i < 2; ++i) {
    const int mbase = m0 + wm * 32 + i * 16 + 8 * lh;
#pragma unroll
    for (int j = 0; j < 2; ++j) {
      const int n = n0 + wn * 32 + j * 16 + frow;
      v8f acc = (i == 0) ? (j == 0 ? acc00 : acc01) : (j == 0 ? acc10 : acc11);
      if (isKey) {
        const float bias = bk[n];
#pragma unroll
        for (int r = 0; r < 8; ++r)
          key_out[(size_t)(mbase + r) * (GG * HH) + n] = acc[r] + bias;
      } else {
        const int nv = n - GG * HH;
        const float bias = bv[nv];
#pragma unroll
        for (int r = 0; r < 8; ++r)
          v_out[(size_t)(mbase + r) * HH + nv] = acc[r] + bias;
      }
    }
  }
}

// ---------------------------------------------------------------------------
// Kernel 2: gate + value (in-place over key buffer) + inv-rms for conv stage.
// ---------------------------------------------------------------------------
__device__ inline void blockReduce3(float& x0, float& x1, float& x2,
                                    float* sm, int tid) {
#pragma unroll
  for (int off = 16; off > 0; off >>= 1) {
    x0 += __shfl_down(x0, off, 32);
    x1 += __shfl_down(x1, off, 32);
    x2 += __shfl_down(x2, off, 32);
  }
  const int w = tid >> 5, l = tid & 31;
  if (l == 0) { sm[w] = x0; sm[8 + w] = x1; sm[16 + w] = x2; }
  __syncthreads();
  if (tid == 0) {
    float a = 0.f, b = 0.f, c = 0.f;
#pragma unroll
    for (int i = 0; i < 8; ++i) { a += sm[i]; b += sm[8 + i]; c += sm[16 + i]; }
    sm[0] = a; sm[8] = b; sm[16] = c;
  }
  __syncthreads();
  x0 = sm[0]; x1 = sm[8]; x2 = sm[16];
  __syncthreads();
}

__global__ __launch_bounds__(256) void gate_value_kernel(
    const float* __restrict__ hidden, const float* __restrict__ w1,
    const float* __restrict__ w2, const float* __restrict__ vbuf,
    float* __restrict__ valbuf, float* __restrict__ inv_rms)
{
  __shared__ float sm[24];
  const int tid = threadIdx.x;
  const size_t m = blockIdx.x;

  const float4 v4 = *(const float4*)(vbuf + m * HH + tid * 4);
  float svv = v4.x * v4.x + v4.y * v4.y + v4.z * v4.z + v4.w * v4.w;
  float z0 = 0.f, z1 = 0.f;
  blockReduce3(svv, z0, z1, sm, tid);
  const float mvv = svv * (1.0f / (float)HH);

#pragma unroll
  for (int g = 0; g < GG; ++g) {
    const size_t off = m * (GG * HH) + (size_t)g * HH + tid * 4;
    const float4 k4 = *(const float4*)(valbuf + off);
    const float4 q4 = *(const float4*)(hidden + off);
    const float4 wa = *(const float4*)(w1 + g * HH + tid * 4);
    const float4 wb = *(const float4*)(w2 + g * HH + tid * 4);
    float skk = k4.x * k4.x + k4.y * k4.y + k4.z * k4.z + k4.w * k4.w;
    float sqq = q4.x * q4.x + q4.y * q4.y + q4.z * q4.z + q4.w * q4.w;
    float skq = k4.x * q4.x * wa.x * wb.x + k4.y * q4.y * wa.y * wb.y
              + k4.z * q4.z * wa.z * wb.z + k4.w * q4.w * wa.w * wb.w;
    blockReduce3(skk, sqq, skq, sm, tid);

    const float gpre = skq * rsqrtf(skk * (1.f / (float)HH) + 1.1920929e-7f)
                           * rsqrtf(sqq * (1.f / (float)HH) + 1.1920929e-7f)
                           * (1.f / 32.f);
    const float sgn = (gpre > 0.f) ? 1.f : ((gpre < 0.f) ? -1.f : 0.f);
    const float gi  = sqrtf(fmaxf(fabsf(gpre), 1e-6f)) * sgn;
    const float gate = 1.f / (1.f + expf(-gi));

    float4 val;
    val.x = gate * v4.x; val.y = gate * v4.y;
    val.z = gate * v4.z; val.w = gate * v4.w;
    *(float4*)(valbuf + off) = val;   // overwrite key with value, in place

    if (tid == 0)
      inv_rms[m * GG + g] = rsqrtf(gate * gate * mvv + 1e-5f);
  }
}

// ---------------------------------------------------------------------------
// Kernel 3: causal dilated depthwise conv (K=4, dil=3) on x_norm + SiLU + res.
// ---------------------------------------------------------------------------
__global__ __launch_bounds__(256) void conv_silu_kernel(
    const float* __restrict__ valbuf, const float* __restrict__ inv_rms,
    const float* __restrict__ wn, const float* __restrict__ convw,
    float* __restrict__ out)
{
  const size_t idx = (size_t)blockIdx.x * 256 + threadIdx.x;   // [M * 4096)
  const int c = (int)(idx & (GG * HH - 1));
  const int m = (int)(idx >> 12);
  const int t = m & (TT - 1);
  const int g = c >> 10;

  const float wnc = wn[c];
  const float4 w4 = *(const float4*)(convw + (size_t)c * 4);
  const float wj[4] = { w4.x, w4.y, w4.z, w4.w };

  const float vself = valbuf[idx];
  float acc = 0.f;
#pragma unroll
  for (int j = 0; j < 4; ++j) {
    const int d = (3 - j) * 3;            // 9, 6, 3, 0
    if (t - d >= 0) {
      const size_t mm = (size_t)(m - d);
      const float x = ((j == 3) ? vself : valbuf[mm * (GG * HH) + c])
                      * inv_rms[mm * GG + g] * wnc;
      acc += wj[j] * x;
    }
  }
  const float y = acc / (1.f + expf(-acc));   // silu
  out[idx] = vself + y;
}

// ---------------------------------------------------------------------------
// Launch.  ws layout (bytes):
//   [0)          value/key  [M,4096] f32   256 MB
//   [+0x10000000) v          [M,1024] f32    64 MB
//   [+0x14000000) inv_rms    [M,4]    f32   256 KB
//   [+0x14040000) embH       [M,1024] f16    32 MB
//   [+0x16040000) WH         [5120,1024] f16 10 MB
// ---------------------------------------------------------------------------
extern "C" void kernel_launch(void* const* d_in, const int* in_sizes, int n_in,
                              void* d_out, int out_size, void* d_ws, size_t ws_size,
                              hipStream_t stream) {
  const float* emb   = (const float*)d_in[0];
  const float* hid   = (const float*)d_in[1];
  const float* Wv    = (const float*)d_in[2];
  const float* bv    = (const float*)d_in[3];
  const float* Wk    = (const float*)d_in[4];
  const float* bk    = (const float*)d_in[5];
  const float* w1    = (const float*)d_in[6];
  const float* w2    = (const float*)d_in[7];
  const float* wn    = (const float*)d_in[8];
  const float* convw = (const float*)d_in[9];
  float* out = (float*)d_out;

  char* ws = (char*)d_ws;
  float* valbuf = (float*)ws;
  float* vbuf   = (float*)(ws + (size_t)MTOT * (GG * HH) * 4);
  float* invb   = (float*)(ws + (size_t)MTOT * (GG * HH) * 4
                              + (size_t)MTOT * HH * 4);
  unsigned short* embH = (unsigned short*)(ws + (size_t)MTOT * (GG * HH) * 4
                                              + (size_t)MTOT * HH * 4
                                              + (size_t)MTOT * GG * 4);
  unsigned short* WH = embH + (size_t)MTOT * EE;

  // f32 -> f16 pre-pass (emb, Wk, Wv -> embH, WH)
  f32_to_f16_kernel<<<(MTOT * EE) / 2048, 256, 0, stream>>>(emb, (unsigned int*)embH);
  f32_to_f16_kernel<<<(GG * HH * EE) / 2048, 256, 0, stream>>>(Wk, (unsigned int*)WH);
  f32_to_f16_kernel<<<(HH * EE) / 2048, 256, 0, stream>>>(
      Wv, (unsigned int*)(WH + (size_t)GG * HH * EE));

  dim3 g1(NTOT / TN, MTOT / TM);   // 40 x 256 blocks
  proj_gemm_kernel<<<g1, 256, 0, stream>>>(embH, WH, bk, bv, valbuf, vbuf);

  gate_value_kernel<<<MTOT, 256, 0, stream>>>(hid, w1, w2, vbuf, valbuf, invb);

  conv_silu_kernel<<<(MTOT * (GG * HH)) / 256, 256, 0, stream>>>(
      valbuf, invb, wn, convw, out);
}
